// GraphletMessagePass_26310969655379
// MI455X (gfx1250) — compile-verified, matrix operand order checked by reading, not run on hardware
//
#include <hip/hip_runtime.h>
#include <math.h>

// ---------------------------------------------------------------------------
// GraphletMessagePass for MI455X (gfx1250): bf16 WMMA GEMM core + elementwise
// glue. All matmuls go through k_gemm_bf16 (v_wmma_f32_16x16x32_bf16, f32
// accumulate); GEMM tiles are staged with GLOBAL_LOAD_ASYNC_TO_LDS_B128
// (ASYNCcnt-tracked direct-to-LDS DMA path), double-buffered in LDS.
// ---------------------------------------------------------------------------

typedef __attribute__((ext_vector_type(16))) __bf16 v16bf;
typedef __attribute__((ext_vector_type(8)))  __bf16 v8bf;
typedef __attribute__((ext_vector_type(8)))  float  v8f;

// Problem constants (match reference).
constexpr int    BB = 8192, NN = 4, TT = 3, EE = 256;
constexpr size_t M1 = (size_t)BB * NN * TT;   // 98304 rows for MLP stages
constexpr size_t GB = (size_t)BB * TT;        // 24576 GRU batch
constexpr size_t RN = (size_t)BB * NN;        // 32768 rows for readout

static __device__ __forceinline__ __bf16 f2bf(float f) {
  union { float f; unsigned u; } v; v.f = f;
  unsigned r = v.u + 0x7FFFu + ((v.u >> 16) & 1u);   // round-to-nearest-even
  union { unsigned short s; __bf16 b; } o;
  o.s = (unsigned short)(r >> 16);
  return o.b;
}
static __device__ __forceinline__ float bf2f(__bf16 b) {
  union { unsigned short s; __bf16 b; } i; i.b = b;
  union { unsigned u; float f; } o; o.u = ((unsigned)i.s) << 16;
  return o.f;
}

// Async direct-to-LDS copy: 16B per lane, global -> LDS, tracked by ASYNCcnt.
static __device__ __forceinline__ void async_ld16(unsigned ldsOff, const void* g) {
  asm volatile("global_load_async_to_lds_b128 %0, %1, off"
               :
               : "v"(ldsOff), "v"(g)
               : "memory");
}
static __device__ __forceinline__ void wait_async0() {
  asm volatile("s_wait_asynccnt 0x0" ::: "memory");
}

// ---------------------------------------------------------------------------
// WMMA fragment load from LDS, per CDNA5 16-bit A/B layout (ISA 7.12.2):
// lane r = row (A) / col (B); lanes 0-15 hold K[0..8)+K[16..24),
// lanes 16-31 hold K[8..16)+K[24..32). Two ds_load_b128 per fragment.
// ---------------------------------------------------------------------------
static __device__ __forceinline__ v16bf frag32(const __bf16* base, int lane) {
  const __bf16* p = base + (size_t)(lane & 15) * 32 + ((lane & 16) ? 8 : 0);
  v8bf lo = *(const v8bf*)p;
  v8bf hi = *(const v8bf*)(p + 16);
  return __builtin_shufflevector(lo, hi, 0,1,2,3,4,5,6,7,8,9,10,11,12,13,14,15);
}

// C = act(A[M,K] x Wt[N,K]^T + bias). 256 threads, 128x128 tile, BK=32,
// double-buffered LDS, 8 waves each computing a 32x64 sub-tile (8 WMMAs/step).
__launch_bounds__(256)
__global__ void k_gemm_bf16(const __bf16* __restrict__ A,
                            const __bf16* __restrict__ Wt,
                            const float*  __restrict__ bias,
                            float*        __restrict__ outF,
                            __bf16*       __restrict__ outB,
                            int M, int Nc, int K, int relu) {
  __shared__ __align__(16) __bf16 lA[2][128 * 32];
  __shared__ __align__(16) __bf16 lB[2][128 * 32];

  const int tid  = threadIdx.x;
  const int lane = tid & 31;
  const int wave = tid >> 5;
  const int wm   = wave & 3;      // 4 waves down M (32 rows each)
  const int wn   = wave >> 2;     // 2 waves across N (64 cols each)
  const size_t blockM = (size_t)blockIdx.y * 128;
  const size_t blockN = (size_t)blockIdx.x * 128;

  // LDS byte offsets of the staging buffers (aperture keeps offset in [31:0]).
  const unsigned baseA = (unsigned)(size_t)&lA[0][0];
  const unsigned baseB = (unsigned)(size_t)&lB[0][0];

  v8f acc[2][4];
#pragma unroll
  for (int i = 0; i < 2; ++i)
#pragma unroll
    for (int j = 0; j < 4; ++j)
#pragma unroll
      for (int v = 0; v < 8; ++v) acc[i][j][v] = 0.0f;

  auto stage = [&](int buf, int k0) {
#pragma unroll
    for (int j = 0; j < 2; ++j) {
      int c   = tid + j * 256;          // 512 chunks of 8 bf16 (16B)
      int row = c >> 2;
      int col = (c & 3) * 8;
      const unsigned lo = (unsigned)(buf * (128 * 32 * 2) + (row * 32 + col) * 2);
      async_ld16(baseA + lo, &A[(blockM + row) * (size_t)K + k0 + col]);
      async_ld16(baseB + lo, &Wt[(blockN + row) * (size_t)K + k0 + col]);
    }
  };

  stage(0, 0);
  wait_async0();
  __syncthreads();
  const int nsteps = K >> 5;
  for (int s = 0; s < nsteps; ++s) {
    const int buf = s & 1;
    if (s + 1 < nsteps) stage(buf ^ 1, (s + 1) << 5);

    const __bf16* Abase = &lA[buf][wm * 32 * 32];
    const __bf16* Bbase = &lB[buf][wn * 64 * 32];
    v16bf af[2], bfr[4];
#pragma unroll
    for (int i = 0; i < 2; ++i) af[i]  = frag32(Abase + i * 16 * 32, lane);
#pragma unroll
    for (int j = 0; j < 4; ++j) bfr[j] = frag32(Bbase + j * 16 * 32, lane);
#pragma unroll
    for (int i = 0; i < 2; ++i)
#pragma unroll
      for (int j = 0; j < 4; ++j)
        acc[i][j] = __builtin_amdgcn_wmma_f32_16x16x32_bf16(
            false, af[i], false, bfr[j], (short)0, acc[i][j], false, false);

    wait_async0();        // next tile fully in LDS before the barrier
    __syncthreads();
  }

  // C/D layout: VGPR v -> M = base + v (+8 for lanes 16-31), N = lane&15.
  const int col0  = (int)blockN + wn * 64;
  const int rbase = (int)blockM + wm * 32 + ((lane & 16) ? 8 : 0);
  const int nlane = lane & 15;
#pragma unroll
  for (int i = 0; i < 2; ++i) {
#pragma unroll
    for (int j = 0; j < 4; ++j) {
      const int coln = col0 + j * 16 + nlane;
      const float bv = bias ? bias[coln] : 0.0f;
#pragma unroll
      for (int v = 0; v < 8; ++v) {
        const size_t row = (size_t)(rbase + i * 16 + v);
        float val = acc[i][j][v] + bv;
        if (relu) val = fmaxf(val, 0.0f);
        const size_t oidx = row * (size_t)Nc + coln;
        if (outF) outF[oidx] = val;
        if (outB) outB[oidx] = f2bf(val);
      }
    }
  }
}

// --------------------------- elementwise kernels ---------------------------

__global__ void k_conv_copy(const float* __restrict__ w, __bf16* __restrict__ o,
                            size_t n) {
  size_t i = (size_t)blockIdx.x * 256 + threadIdx.x;
  if (i < n) o[i] = f2bf(w[i]);
}

__global__ void k_conv_transpose(const float* __restrict__ w,  // (K,Nc)
                                 __bf16* __restrict__ o,       // (Nc,K)
                                 int K, int Nc, size_t n) {
  size_t i = (size_t)blockIdx.x * 256 + threadIdx.x;
  if (i >= n) return;
  int col = (int)(i % Nc);
  int k   = (int)(i / Nc);
  o[(size_t)col * K + k] = f2bf(w[i]);
}

__global__ void k_mean_f32(const float* __restrict__ ge, __bf16* __restrict__ msg) {
  size_t i = (size_t)blockIdx.x * 256 + threadIdx.x;       // over RN*EE
  if (i >= RN * EE) return;
  size_t r = i / EE, e = i % EE;
  float s = 0.0f;
  for (int t = 0; t < TT; ++t) s += ge[(r * TT + t) * EE + e];
  msg[i] = f2bf(s * (1.0f / 3.0f));
}

__global__ void k_mean_bf16(const __bf16* __restrict__ ge, __bf16* __restrict__ msg) {
  size_t i = (size_t)blockIdx.x * 256 + threadIdx.x;
  if (i >= RN * EE) return;
  size_t r = i / EE, e = i % EE;
  float s = 0.0f;
  for (int t = 0; t < TT; ++t) s += bf2f(ge[(r * TT + t) * EE + e]);
  msg[i] = f2bf(s * (1.0f / 3.0f));
}

// cat = [init(f32->bf16) | msg] per (b,n,t) row
__global__ void k_pack_cat1(const float* __restrict__ init,
                            const __bf16* __restrict__ msg,
                            __bf16* __restrict__ cat) {
  size_t i = (size_t)blockIdx.x * 256 + threadIdx.x;       // over M1*512
  if (i >= M1 * 512) return;
  size_t row = i >> 9; int c = (int)(i & 511);
  size_t rn  = row / TT;
  cat[i] = (c < 256) ? f2bf(init[row * EE + c]) : msg[rn * EE + (c - 256)];
}

// cat = [ge(bf16) | init(f32->bf16)]
__global__ void k_pack_cat2(const __bf16* __restrict__ gebf,
                            const float* __restrict__ init,
                            __bf16* __restrict__ cat) {
  size_t i = (size_t)blockIdx.x * 256 + threadIdx.x;
  if (i >= M1 * 512) return;
  size_t row = i >> 9; int c = (int)(i & 511);
  cat[i] = (c < 256) ? gebf[row * 256 + c] : f2bf(init[row * EE + (c - 256)]);
}

// torch "raw reshape": Xseq[s][t*B+b2][e] = E1[(s*B+b2)*T + t][e]
__global__ void k_pack_seq(const __bf16* __restrict__ e1, __bf16* __restrict__ xs) {
  size_t i = (size_t)blockIdx.x * 256 + threadIdx.x;       // over 4*GB*EE
  if (i >= 4 * GB * EE) return;
  size_t e = i % EE;
  size_t m = (i / EE) % GB;
  size_t s = i / (EE * GB);
  size_t t = m / BB, b2 = m % BB;
  xs[i] = e1[((s * BB + b2) * TT + t) * EE + e];
}

// inverse: Cat[(b*N+n)*T + t][c] = Y[r/B][t*B + r%B][c],  r = b*N+n
__global__ void k_unpack_y(const __bf16* __restrict__ y, __bf16* __restrict__ cat) {
  size_t i = (size_t)blockIdx.x * 256 + threadIdx.x;       // over M1*512
  if (i >= M1 * 512) return;
  size_t row = i >> 9; int c = (int)(i & 511);
  size_t rn = row / TT;  int t = (int)(row % TT);
  size_t s = rn >> 13, b2 = rn & (BB - 1);
  cat[i] = y[(s * GB + (size_t)t * BB + b2) * 512 + c];
}

__global__ void k_h0(const __bf16* __restrict__ xslice, float* __restrict__ h,
                     __bf16* __restrict__ hbf) {
  size_t i = (size_t)blockIdx.x * 256 + threadIdx.x;       // over GB*EE
  if (i >= GB * EE) return;
  h[i] = bf2f(xslice[i]);
  hbf[i] = xslice[i];
}

// GRU gates (biases already folded into gi/gh via GEMM epilogue)
__global__ void k_gru_gate(const float* __restrict__ gi, const float* __restrict__ gh,
                           float* __restrict__ h, __bf16* __restrict__ hbf,
                           __bf16* __restrict__ yout /* row stride 512 */) {
  size_t i = (size_t)blockIdx.x * 256 + threadIdx.x;       // over GB*EE
  if (i >= GB * EE) return;
  size_t m = i / EE; int e = (int)(i % EE);
  const float* gim = gi + m * 768;
  const float* ghm = gh + m * 768;
  float r = 1.0f / (1.0f + __expf(-(gim[e]       + ghm[e])));
  float z = 1.0f / (1.0f + __expf(-(gim[256 + e] + ghm[256 + e])));
  float nv = tanhf(gim[512 + e] + r * ghm[512 + e]);
  float h2 = (1.0f - z) * nv + z * h[i];
  h[i] = h2;
  __bf16 hb = f2bf(h2);
  hbf[i] = hb;
  yout[m * 512 + e] = hb;
}

__global__ void k_prod(const __bf16* __restrict__ a, const __bf16* __restrict__ b,
                       __bf16* __restrict__ o) {
  size_t i = (size_t)blockIdx.x * 256 + threadIdx.x;       // over M1*256
  if (i >= M1 * 256) return;
  o[i] = f2bf(bf2f(a[i]) * bf2f(b[i]));
}

__global__ void k_sum_T(const float* __restrict__ f5, __bf16* __restrict__ emb) {
  size_t i = (size_t)blockIdx.x * 256 + threadIdx.x;       // over RN*EE
  if (i >= RN * EE) return;
  size_t r = i / EE, e = i % EE;
  float s = 0.0f;
  for (int t = 0; t < TT; ++t) s += f5[(r * TT + t) * EE + e];
  emb[i] = f2bf(s);
}

// ---------------------------------------------------------------------------

static inline unsigned eb(size_t n) { return (unsigned)((n + 255) / 256); }

extern "C" void kernel_launch(void* const* d_in, const int* in_sizes, int n_in,
                              void* d_out, int out_size, void* d_ws, size_t ws_size,
                              hipStream_t stream) {
  (void)in_sizes; (void)n_in; (void)out_size; (void)ws_size;
  // JAX pytree flattening: sorted dict keys.
  // 0: graphlet_embed; then params: fo2,fr2,fr4,fr5,fu1,fu3 (W1,W2,b1,b2 each),
  // then gru[0],gru[1] (Whh,Wih,bhh,bih each).
  const float* ge = (const float*)d_in[0];
  const float *mW1[6], *mW2[6], *mb1[6], *mb2[6];
  for (int m = 0; m < 6; ++m) {                 // fo2, fr2, fr4, fr5, fu1, fu3
    mW1[m] = (const float*)d_in[1 + 4 * m + 0];
    mW2[m] = (const float*)d_in[1 + 4 * m + 1];
    mb1[m] = (const float*)d_in[1 + 4 * m + 2];
    mb2[m] = (const float*)d_in[1 + 4 * m + 3];
  }
  enum { FO2 = 0, FR2 = 1, FR4 = 2, FR5 = 3, FU1 = 4, FU3 = 5 };
  const float* gWhh[2] = { (const float*)d_in[25], (const float*)d_in[29] };
  const float* gWih[2] = { (const float*)d_in[26], (const float*)d_in[30] };
  const float* gbhh[2] = { (const float*)d_in[27], (const float*)d_in[31] };
  const float* gbih[2] = { (const float*)d_in[28], (const float*)d_in[32] };

  // Bump allocator over d_ws (total ~985 MB).
  char* ws = (char*)d_ws;
  size_t off = 0;
  auto alloc = [&](size_t bytes) {
    void* p = ws + off;
    off += (bytes + 255) & ~(size_t)255;
    return p;
  };
  const int mlpK[6] = { 256, 256, 512, 256, 512, 512 };   // W1 input dims
  __bf16 *w1t[6], *w2t[6];
  for (int m = 0; m < 6; ++m) {
    w1t[m] = (__bf16*)alloc((size_t)256 * mlpK[m] * 2);
    w2t[m] = (__bf16*)alloc((size_t)256 * 256 * 2);
  }
  const int gruDin[2] = { 256, 512 };
  __bf16 *wihB[2], *whhB[2];
  for (int l = 0; l < 2; ++l) {
    wihB[l] = (__bf16*)alloc((size_t)2 * 768 * gruDin[l] * 2);
    whhB[l] = (__bf16*)alloc((size_t)2 * 768 * 256 * 2);
  }
  __bf16* msg  = (__bf16*)alloc(RN * EE * 2);
  __bf16* Acat = (__bf16*)alloc(M1 * 512 * 2);
  __bf16* Ahid = (__bf16*)alloc(M1 * 256 * 2);
  __bf16* E1bf = (__bf16*)alloc(M1 * 256 * 2);
  __bf16* Xseq = (__bf16*)alloc(4 * GB * EE * 2);
  __bf16* Y0   = (__bf16*)alloc(4 * GB * 512 * 2);
  __bf16* Y1   = (__bf16*)alloc(4 * GB * 512 * 2);
  float*  hS   = (float*)alloc(GB * EE * 4);
  __bf16* hBF  = (__bf16*)alloc(GB * EE * 2);
  float*  Gi   = (float*)alloc(GB * 768 * 4);
  float*  Gh   = (float*)alloc(GB * 768 * 4);
  __bf16* Gebf = (__bf16*)alloc(M1 * 256 * 2);
  __bf16* e1b  = (__bf16*)alloc(M1 * 256 * 2);
  __bf16* e2b  = (__bf16*)alloc(M1 * 256 * 2);
  __bf16* Pbuf = (__bf16*)alloc(M1 * 256 * 2);
  float*  F5   = (float*)alloc(M1 * 256 * 4);
  __bf16* embb = (__bf16*)alloc(RN * EE * 2);

  auto gemm = [&](const __bf16* A, const __bf16* Wt, const float* bias,
                  float* oF, __bf16* oB, int M, int Nc, int K, int relu) {
    dim3 g(Nc / 128, M / 128);
    k_gemm_bf16<<<g, dim3(256), 0, stream>>>(A, Wt, bias, oF, oB, M, Nc, K, relu);
  };

  // ---- convert weights to bf16 (MLPs transposed to (N,K); GRU already (N,K))
  for (int m = 0; m < 6; ++m) {
    size_t n1 = (size_t)mlpK[m] * 256;
    k_conv_transpose<<<eb(n1), 256, 0, stream>>>(mW1[m], w1t[m], mlpK[m], 256, n1);
    size_t n2 = 256 * 256;
    k_conv_transpose<<<eb(n2), 256, 0, stream>>>(mW2[m], w2t[m], 256, 256, n2);
  }
  for (int l = 0; l < 2; ++l) {
    size_t ni = (size_t)2 * 768 * gruDin[l];
    k_conv_copy<<<eb(ni), 256, 0, stream>>>(gWih[l], wihB[l], ni);
    size_t nh = (size_t)2 * 768 * 256;
    k_conv_copy<<<eb(nh), 256, 0, stream>>>(gWhh[l], whhB[l], nh);
  }

  for (int pass = 0; pass < 2; ++pass) {
    // msg = mean over T (pass0 from original ge, pass1 from updated ge)
    if (pass == 0) k_mean_f32 <<<eb(RN * EE), 256, 0, stream>>>(ge, msg);
    else           k_mean_bf16<<<eb(RN * EE), 256, 0, stream>>>(Gebf, msg);

    // fu1: MLP(concat(init, msg))
    k_pack_cat1<<<eb(M1 * 512), 256, 0, stream>>>(ge, msg, Acat);
    gemm(Acat, w1t[FU1], mb1[FU1], nullptr, Ahid, (int)M1, 256, 512, 1);
    gemm(Ahid, w2t[FU1], mb2[FU1], nullptr, E1bf, (int)M1, 256, 256, 0);

    // reshape into GRU sequence (seq=4, batch=24576)
    k_pack_seq<<<eb(4 * GB * EE), 256, 0, stream>>>(E1bf, Xseq);

    // bidirectional 2-layer GRU, h0 = layer-0 input slices
    for (int l = 0; l < 2; ++l) {
      const __bf16* Ain = (l == 0) ? Xseq : Y0;
      __bf16* Yout = (l == 0) ? Y0 : Y1;
      const int din = gruDin[l];
      for (int d = 0; d < 2; ++d) {
        k_h0<<<eb(GB * EE), 256, 0, stream>>>(Xseq + (size_t)(2 * l + d) * GB * EE,
                                              hS, hBF);
        for (int k = 0; k < 4; ++k) {
          const int s = d ? (3 - k) : k;
          gemm(Ain + (size_t)s * GB * din, wihB[l] + (size_t)d * 768 * din,
               gbih[l] + d * 768, Gi, nullptr, (int)GB, 768, din, 0);
          gemm(hBF, whhB[l] + (size_t)d * 768 * 256,
               gbhh[l] + d * 768, Gh, nullptr, (int)GB, 768, 256, 0);
          k_gru_gate<<<eb(GB * EE), 256, 0, stream>>>(
              Gi, Gh, hS, hBF, Yout + (size_t)s * GB * 512 + d * 256);
        }
      }
    }

    // fu3 on reshaped GRU output
    k_unpack_y<<<eb(M1 * 512), 256, 0, stream>>>(Y1, Acat);
    gemm(Acat, w1t[FU3], mb1[FU3], nullptr, Ahid, (int)M1, 256, 512, 1);
    gemm(Ahid, w2t[FU3], mb2[FU3], nullptr, Gebf, (int)M1, 256, 256, 0);

    // readout: e1 = fr4(concat(ge, init)); e2 = fr2(ge); emb = sum_T fr5(e1*e2)
    k_pack_cat2<<<eb(M1 * 512), 256, 0, stream>>>(Gebf, ge, Acat);
    gemm(Acat, w1t[FR4], mb1[FR4], nullptr, Ahid, (int)M1, 256, 512, 1);
    gemm(Ahid, w2t[FR4], mb2[FR4], nullptr, e1b, (int)M1, 256, 256, 0);
    gemm(Gebf, w1t[FR2], mb1[FR2], nullptr, Ahid, (int)M1, 256, 256, 1);
    gemm(Ahid, w2t[FR2], mb2[FR2], nullptr, e2b, (int)M1, 256, 256, 0);
    k_prod<<<eb(M1 * 256), 256, 0, stream>>>(e1b, e2b, Pbuf);
    gemm(Pbuf, w1t[FR5], mb1[FR5], nullptr, Ahid, (int)M1, 256, 256, 1);
    gemm(Ahid, w2t[FR5], mb2[FR5], F5, nullptr, (int)M1, 256, 256, 0);
    k_sum_T<<<eb(RN * EE), 256, 0, stream>>>(F5, embb);
  }

  // final output MLP fo2 -> d_out (B,N,OUT) f32
  gemm(embb, w1t[FO2], mb1[FO2], nullptr, Ahid, (int)RN, 256, 256, 1);
  gemm(Ahid, w2t[FO2], mb2[FO2], (float*)d_out, nullptr, (int)RN, 256, 256, 0);
}
